// GPT_58471684767879
// MI455X (gfx1250) — compile-verified, matrix operand order checked by reading, not run on hardware
//
#include <hip/hip_runtime.h>
#include <math.h>
#include <stdint.h>

// ---------------- model constants ----------------
#define HDIM   1024
#define NHEAD  16
#define KVHN   4
#define QPKN   4          // NHEAD / KVHN
#define HDH    64         // head dim
#define KVDIM  256        // KVHN * HDH
#define FFD    4096
#define VOCAB  32000
#define BBAT   2
#define SSEQ   2048
#define MTOK   (BBAT * SSEQ)   // 4096 rows in every GEMM

typedef __attribute__((ext_vector_type(16))) __bf16 v16bf;
typedef __attribute__((ext_vector_type(8)))  __bf16 bf16x8;
typedef __attribute__((ext_vector_type(8)))  float  v8f;
typedef __attribute__((ext_vector_type(4)))  float  fltx4;

// TDM descriptor vector types (clang-23 6-arg builtin signature:
// (uint32x4 g0, int32x8 g1, int32x4 g2, int32x4 g3, int32x8 extra, i32 cpol))
typedef __attribute__((ext_vector_type(4))) unsigned int tdm_u4;
typedef __attribute__((ext_vector_type(8))) int          tdm_i8;
typedef __attribute__((ext_vector_type(4))) int          tdm_i4;

#if __has_builtin(__builtin_amdgcn_tensor_load_to_lds)
#define HAVE_TDM 1
#else
#define HAVE_TDM 0
#endif

static __device__ inline v16bf make_frag(const __bf16* lo, const __bf16* hi) {
    bf16x8 a = *(const bf16x8*)lo;
    bf16x8 b = *(const bf16x8*)hi;
    v16bf r;
#pragma unroll
    for (int i = 0; i < 8; ++i) { r[i] = a[i]; r[8 + i] = b[i]; }
    return r;
}

// ---------------- embedding: x = tok_emb[ids] + pos_emb ----------------
__global__ __launch_bounds__(256) void embed_kernel(const int* __restrict__ ids,
                                                    const float* __restrict__ tok,
                                                    const float* __restrict__ pos,
                                                    float* __restrict__ x) {
    long i = (long)blockIdx.x * 256 + threadIdx.x;   // over B*S*H
    int t = (int)(i >> 10);          // token index (H = 1024)
    int c = (int)(i & (HDIM - 1));
    int s = t & (SSEQ - 1);
    int id = ids[t];
    x[i] = tok[(long)id * HDIM + c] + pos[(long)s * HDIM + c];
}

// ---------------- layernorm (fp32 in, bf16 out), one block per row ----------------
__global__ __launch_bounds__(256) void ln_kernel(const float* __restrict__ x,
                                                 const float* __restrict__ g,
                                                 const float* __restrict__ bta,
                                                 __bf16* __restrict__ out) {
    int row = blockIdx.x;
    const float* xr = x + (long)row * HDIM;
    __shared__ float red[256];
    float v[4];
    float s = 0.f;
#pragma unroll
    for (int i = 0; i < 4; ++i) { v[i] = xr[threadIdx.x + i * 256]; s += v[i]; }
    red[threadIdx.x] = s; __syncthreads();
    for (int o = 128; o > 0; o >>= 1) {
        if ((int)threadIdx.x < o) red[threadIdx.x] += red[threadIdx.x + o];
        __syncthreads();
    }
    float mean = red[0] * (1.0f / HDIM);
    __syncthreads();
    s = 0.f;
#pragma unroll
    for (int i = 0; i < 4; ++i) { float d = v[i] - mean; s += d * d; }
    red[threadIdx.x] = s; __syncthreads();
    for (int o = 128; o > 0; o >>= 1) {
        if ((int)threadIdx.x < o) red[threadIdx.x] += red[threadIdx.x + o];
        __syncthreads();
    }
    float rs = rsqrtf(red[0] * (1.0f / HDIM) + 1e-5f);
#pragma unroll
    for (int i = 0; i < 4; ++i) {
        int c = threadIdx.x + i * 256;
        out[(long)row * HDIM + c] = (__bf16)((v[i] - mean) * rs * g[c] + bta[c]);
    }
}

// ---------------- WMMA GEMM: C[M,N] = A[M,K](bf16) @ Bw[K,N](fp32->bf16 in LDS) ----------------
// A tile is DMA'd into LDS by the Tensor Data Mover (wave 0 issues, TENSORcnt + barrier
// publishes); B tile is read fp32 (L2-resident weights), converted to bf16 while the DMA runs.
// MODE 0: +bias -> bf16 out.  MODE 1: +bias, Cf += (residual, N==HDIM).
// MODE 2: +bias, exact GELU -> bf16 out.  MODE 3: plain fp32 out (LM head).
template <int MODE>
__global__ __launch_bounds__(256) void gemm_kernel(const __bf16* __restrict__ A,
                                                   const float* __restrict__ Bw,
                                                   const float* __restrict__ bias,
                                                   float* __restrict__ Cf,
                                                   __bf16* __restrict__ Cb,
                                                   int N, int K) {
    __shared__ __bf16 As[128 * 32];   // [m][k], k contiguous (matches TDM row-major fill)
    __shared__ __bf16 Bs[128 * 32];   // [n][k], k contiguous
    const int tid  = threadIdx.x;
    const int lane = tid & 31, w = tid >> 5;
    const int wm = w & 3, wn = w >> 2;          // 4 waves along M, 2 along N
    const int bm0 = blockIdx.y * 128, bn0 = blockIdx.x * 128;
    const int rq = lane & 15, hl = lane >> 4, kb = hl * 8;

    v8f c[2][4];
#pragma unroll
    for (int mt = 0; mt < 2; ++mt)
#pragma unroll
        for (int nt = 0; nt < 4; ++nt)
            c[mt][nt] = (v8f){0.f, 0.f, 0.f, 0.f, 0.f, 0.f, 0.f, 0.f};

    for (int k0 = 0; k0 < K; k0 += 32) {
#if HAVE_TDM
        // ---- A tile via Tensor Data Mover (async DMA, TENSORcnt) ----
        if (w == 0) {
            unsigned long long ga =
                (unsigned long long)(uintptr_t)((const char*)A + ((long)bm0 * K + k0) * 2);
            unsigned ldso = (unsigned)(uintptr_t)(&As[0]);   // LDS byte offset (low 32 bits)
            tdm_u4 g0;
            g0[0] = 1u;                                      // count=1, user mode, no gather
            g0[1] = ldso;                                    // lds_addr
            g0[2] = (unsigned)(ga & 0xffffffffull);          // global_addr[31:0]
            g0[3] = (unsigned)((ga >> 32) & 0x01ffffffull)   // global_addr[56:32]
                    | 0x80000000u;                           // type=2 ("image") in bits[127:126]
            tdm_i8 g1;
            g1[0] = 0x00010000;                              // data_size=1 (2 bytes/elem)
            g1[1] = (int)((K & 0xffff) << 16);               // tensor_dim0 low16 (elems/row)
            g1[2] = (int)(((K >> 16) & 0xffff) |             // tensor_dim0 hi16
                          ((MTOK & 0xffff) << 16));          // tensor_dim1 low16 (rows)
            g1[3] = (int)(32 << 16);                         // tensor_dim1 hi16=0 | tile_dim0=32
            g1[4] = 128;                                     // tile_dim1=128 | tile_dim2=0
            g1[5] = K;                                       // tensor_dim0_stride[31:0]
            g1[6] = 0;                                       // stride hi | dim1_stride lo (unused, 2D)
            g1[7] = 0;
            tdm_i4 z4 = {0, 0, 0, 0};                        // groups 2/3: NULL (<=2D tensor)
            tdm_i8 z8 = {0, 0, 0, 0, 0, 0, 0, 0};            // extra descriptor words (unused)
            __builtin_amdgcn_tensor_load_to_lds(g0, g1, z4, z4, z8, 0);
        }
#else
        // fallback staging (host pass / toolchains without the TDM builtin)
#pragma unroll
        for (int j = 0; j < 2; ++j) {
            int idx = tid + j * 256;
            int row = idx >> 2, kv = (idx & 3) * 8;
            *(bf16x8*)&As[row * 32 + kv] =
                *(const bf16x8*)&A[(long)(bm0 + row) * K + k0 + kv];
        }
#endif
        // ---- B tile: fp32 global read (coalesced b128), convert, transpose into [n][k] ----
#pragma unroll
        for (int j = 0; j < 4; ++j) {
            int idx = tid + j * 256;           // 1024 float4s
            int kk = idx >> 5, n4 = (idx & 31) * 4;
            fltx4 bv = *(const fltx4*)&Bw[(long)(k0 + kk) * N + bn0 + n4];
#pragma unroll
            for (int e = 0; e < 4; ++e) Bs[(n4 + e) * 32 + kk] = (__bf16)bv[e];
        }
        if (k0 + 32 < K)                        // global_prefetch_b8 for next B tile
            __builtin_prefetch(&Bw[(long)(k0 + 32 + (tid >> 3)) * N + bn0 + (tid & 7) * 16], 0, 1);
#if HAVE_TDM
        if (w == 0) __builtin_amdgcn_s_wait_tensorcnt(0);    // DMA landed (issuing wave only)
#endif
        __syncthreads();                         // publish As (TDM) + Bs to all waves

        v16bf af[2];
#pragma unroll
        for (int mt = 0; mt < 2; ++mt) {
            int row = wm * 32 + mt * 16 + rq;
            af[mt] = make_frag(&As[row * 32 + kb], &As[row * 32 + kb + 16]);
        }
#pragma unroll
        for (int nt = 0; nt < 4; ++nt) {
            int col = wn * 64 + nt * 16 + rq;
            v16bf bf = make_frag(&Bs[col * 32 + kb], &Bs[col * 32 + kb + 16]);
#pragma unroll
            for (int mt = 0; mt < 2; ++mt)
                c[mt][nt] = __builtin_amdgcn_wmma_f32_16x16x32_bf16(
                    false, af[mt], false, bf, (short)0, c[mt][nt], false, false);
        }
        __syncthreads();                         // all waves done with tiles before restage
    }

    // epilogue (C layout: row = r + 8*hl, col = lane&15 within 16x16 tile)
#pragma unroll
    for (int mt = 0; mt < 2; ++mt)
#pragma unroll
        for (int nt = 0; nt < 4; ++nt)
#pragma unroll
            for (int r = 0; r < 8; ++r) {
                int rg = bm0 + wm * 32 + mt * 16 + r + hl * 8;
                int cg = bn0 + wn * 64 + nt * 16 + rq;
                float v = c[mt][nt][r];
                if (MODE != 3) v += bias[cg];
                long o = (long)rg * N + cg;
                if (MODE == 0) {
                    Cb[o] = (__bf16)v;
                } else if (MODE == 1) {
                    Cf[o] += v;                       // residual accumulate (N == HDIM)
                } else if (MODE == 2) {
                    v = 0.5f * v * (1.0f + erff(v * 0.70710678118f));
                    Cb[o] = (__bf16)v;
                } else {
                    Cf[o] = v;                        // logits
                }
            }
}

// ---------------- flash attention (GQA), bf16 WMMA, fp32 online softmax ----------------
// grid: (S/128, B*NHEAD); block 256 = 8 waves; each wave owns 16 query rows.
__global__ __launch_bounds__(256) void attn_kernel(const __bf16* __restrict__ Q,
                                                   const __bf16* __restrict__ Kx,
                                                   const __bf16* __restrict__ Vx,
                                                   const float* __restrict__ mask,
                                                   __bf16* __restrict__ O) {
    __shared__ float  sc[8][16][64];     // per-wave score tile (fp32)
    __shared__ __bf16 pb[8][16][64];     // per-wave P tile (bf16)
    __shared__ __bf16 vs[64][64];        // V tile transposed: [d][key]
    __shared__ float  rstat[8][16];      // per-row broadcast (rescale / inv-sum)

    const int tid = threadIdx.x, lane = tid & 31, w = tid >> 5;
    const int rq = lane & 15, hl = lane >> 4, kb = hl * 8;
    const int bh = blockIdx.y;
    const int b = bh >> 4, h = bh & 15, g = h >> 2;   // kv head = h / QPKN
    const int qrow0 = blockIdx.x * 128 + w * 16;

    const __bf16* Qp = Q + (long)b * SSEQ * HDIM + (long)h * HDH;
    const __bf16* Kp = Kx + (long)b * SSEQ * KVDIM + (long)g * HDH;
    const __bf16* Vp = Vx + (long)b * SSEQ * KVDIM + (long)g * HDH;

    // Q fragments for this wave's 16 rows (HD=64 -> two K=32 chunks), straight from global
    v16bf aq[2];
#pragma unroll
    for (int kc = 0; kc < 2; ++kc) {
        const __bf16* p = &Qp[(long)(qrow0 + rq) * HDIM + kc * 32 + kb];
        aq[kc] = make_frag(p, p + 16);
    }

    v8f co[4];
#pragma unroll
    for (int nt = 0; nt < 4; ++nt) co[nt] = (v8f){0.f,0.f,0.f,0.f,0.f,0.f,0.f,0.f};
    float m_run = -1e30f, s_run = 0.f;
    const float scale = 0.125f;   // 1/sqrt(64)

    for (int kt = 0; kt < SSEQ / 64; ++kt) {
        __syncthreads();                      // all waves done reading previous vs
        // stage V tile transposed into LDS
#pragma unroll
        for (int j = 0; j < 2; ++j) {
            int idx = tid + j * 256;          // 512 vectors of 8 bf16
            int key = idx >> 3, dv = (idx & 7) * 8;
            bf16x8 vv = *(const bf16x8*)&Vp[(long)(kt * 64 + key) * KVDIM + dv];
#pragma unroll
            for (int e = 0; e < 8; ++e) vs[dv + e][key] = vv[e];
        }
        __syncthreads();

        // S = Q @ K^T : B-fragments loaded directly from global K (row-major by key)
#pragma unroll
        for (int nt = 0; nt < 4; ++nt) {
            v8f cs = (v8f){0.f,0.f,0.f,0.f,0.f,0.f,0.f,0.f};
#pragma unroll
            for (int kc = 0; kc < 2; ++kc) {
                const __bf16* p = &Kp[(long)(kt * 64 + nt * 16 + rq) * KVDIM + kc * 32 + kb];
                v16bf bk = make_frag(p, p + 16);
                cs = __builtin_amdgcn_wmma_f32_16x16x32_bf16(
                    false, aq[kc], false, bk, (short)0, cs, false, false);
            }
#pragma unroll
            for (int r = 0; r < 8; ++r) sc[w][r + hl * 8][nt * 16 + rq] = cs[r];
        }

        // online softmax: lane handles row rq, half hl of the 64 keys
        float lm = -1e30f;
#pragma unroll
        for (int c2 = 0; c2 < 32; ++c2) {
            float t = sc[w][rq][hl * 32 + c2] * scale +
                      mask[(long)b * SSEQ + kt * 64 + hl * 32 + c2];
            sc[w][rq][hl * 32 + c2] = t;
            lm = fmaxf(lm, t);
        }
        lm = fmaxf(lm, __shfl_xor(lm, 16, 32));
        float nm = fmaxf(m_run, lm);
        float al = __expf(m_run - nm);
        float ls = 0.f;
#pragma unroll
        for (int c2 = 0; c2 < 32; ++c2) {
            float p = __expf(sc[w][rq][hl * 32 + c2] - nm);
            pb[w][rq][hl * 32 + c2] = (__bf16)p;
            ls += p;
        }
        ls += __shfl_xor(ls, 16, 32);
        s_run = s_run * al + ls;
        m_run = nm;

        // rescale accumulators: broadcast per-row factor via LDS
        if (hl == 0) rstat[w][rq] = al;
        float alr[8];
#pragma unroll
        for (int r = 0; r < 8; ++r) alr[r] = rstat[w][r + hl * 8];
#pragma unroll
        for (int nt = 0; nt < 4; ++nt)
#pragma unroll
            for (int r = 0; r < 8; ++r) co[nt][r] *= alr[r];

        // O += P @ V
#pragma unroll
        for (int nt = 0; nt < 4; ++nt) {
#pragma unroll
            for (int kc = 0; kc < 2; ++kc) {
                const __bf16* pp = &pb[w][rq][kc * 32 + kb];
                v16bf ap = make_frag(pp, pp + 16);
                const __bf16* vp2 = &vs[nt * 16 + rq][kc * 32 + kb];
                v16bf bv2 = make_frag(vp2, vp2 + 16);
                co[nt] = __builtin_amdgcn_wmma_f32_16x16x32_bf16(
                    false, ap, false, bv2, (short)0, co[nt], false, false);
            }
        }
    }

    // finalize: divide by row sums, store bf16 into [b, s, H] at this head's channels
    float inv = 1.0f / s_run;
    if (hl == 0) rstat[w][rq] = inv;
    float ir[8];
#pragma unroll
    for (int r = 0; r < 8; ++r) ir[r] = rstat[w][r + hl * 8];
#pragma unroll
    for (int nt = 0; nt < 4; ++nt)
#pragma unroll
        for (int r = 0; r < 8; ++r) {
            long row = (long)b * SSEQ + qrow0 + r + hl * 8;
            O[row * HDIM + h * HDH + nt * 16 + rq] = (__bf16)(co[nt][r] * ir[r]);
        }
}

// ---------------- orchestration ----------------
extern "C" void kernel_launch(void* const* d_in, const int* in_sizes, int n_in,
                              void* d_out, int out_size, void* d_ws, size_t ws_size,
                              hipStream_t stream) {
    const int*   ids   = (const int*)d_in[0];
    const float* amask = (const float*)d_in[1];
    const float* tok   = (const float*)d_in[2];
    const float* pos   = (const float*)d_in[3];
    const float* Wq = (const float*)d_in[4],  * bq = (const float*)d_in[5];
    const float* Wk = (const float*)d_in[6],  * bk = (const float*)d_in[7];
    const float* Wv = (const float*)d_in[8],  * bv = (const float*)d_in[9];
    const float* Wo = (const float*)d_in[10], * bo = (const float*)d_in[11];
    const float* W1 = (const float*)d_in[12], * b1 = (const float*)d_in[13];
    const float* W2 = (const float*)d_in[14], * b2 = (const float*)d_in[15];
    const float* ln1g = (const float*)d_in[16], * ln1b = (const float*)d_in[17];
    const float* ln2g = (const float*)d_in[18], * ln2b = (const float*)d_in[19];
    const float* lnfg = (const float*)d_in[20], * lnfb = (const float*)d_in[21];
    const float* Whd  = (const float*)d_in[22];
    float* out = (float*)d_out;

    // workspace layout (all power-of-two sizes, 256B aligned)
    char* ws = (char*)d_ws;
    float*  xf = (float*)ws;                               // 4096*1024 fp32 = 16 MiB
    __bf16* hb = (__bf16*)(ws + 16777216);                 // LN out bf16, 8 MiB
    __bf16* qb = (__bf16*)(ws + 25165824);                 // Q bf16, 8 MiB
    __bf16* kbw = (__bf16*)(ws + 33554432);                // K bf16, 2 MiB
    __bf16* vbw = (__bf16*)(ws + 35651584);                // V bf16, 2 MiB
    __bf16* ob  = (__bf16*)(ws + 37748736);                // attn out bf16, 8 MiB
    __bf16* fb  = (__bf16*)(ws + 46137344);                // FF mid bf16, 32 MiB

    const dim3 blk(256);
    const dim3 gM(MTOK);                                   // one block per LN row

    embed_kernel<<<(MTOK * HDIM) / 256, blk, 0, stream>>>(ids, tok, pos, xf);

    for (int l = 0; l < 4; ++l) {
        ln_kernel<<<gM, blk, 0, stream>>>(xf, ln1g + (long)l * HDIM, ln1b + (long)l * HDIM, hb);

        gemm_kernel<0><<<dim3(HDIM / 128, MTOK / 128), blk, 0, stream>>>(
            hb, Wq + (long)l * HDIM * HDIM, bq + (long)l * HDIM, nullptr, qb, HDIM, HDIM);
        gemm_kernel<0><<<dim3(KVDIM / 128, MTOK / 128), blk, 0, stream>>>(
            hb, Wk + (long)l * HDIM * KVDIM, bk + (long)l * KVDIM, nullptr, kbw, KVDIM, HDIM);
        gemm_kernel<0><<<dim3(KVDIM / 128, MTOK / 128), blk, 0, stream>>>(
            hb, Wv + (long)l * HDIM * KVDIM, bv + (long)l * KVDIM, nullptr, vbw, KVDIM, HDIM);

        attn_kernel<<<dim3(SSEQ / 128, BBAT * NHEAD), blk, 0, stream>>>(qb, kbw, vbw, amask, ob);

        gemm_kernel<1><<<dim3(HDIM / 128, MTOK / 128), blk, 0, stream>>>(
            ob, Wo + (long)l * HDIM * HDIM, bo + (long)l * HDIM, xf, nullptr, HDIM, HDIM);

        ln_kernel<<<gM, blk, 0, stream>>>(xf, ln2g + (long)l * HDIM, ln2b + (long)l * HDIM, hb);

        gemm_kernel<2><<<dim3(FFD / 128, MTOK / 128), blk, 0, stream>>>(
            hb, W1 + (long)l * HDIM * FFD, b1 + (long)l * FFD, nullptr, fb, FFD, HDIM);
        gemm_kernel<1><<<dim3(HDIM / 128, MTOK / 128), blk, 0, stream>>>(
            fb, W2 + (long)l * FFD * HDIM, b2 + (long)l * HDIM, xf, nullptr, HDIM, FFD);
    }

    ln_kernel<<<gM, blk, 0, stream>>>(xf, lnfg, lnfb, hb);
    gemm_kernel<3><<<dim3(VOCAB / 128, MTOK / 128), blk, 0, stream>>>(
        hb, Whd, nullptr, out, nullptr, VOCAB, HDIM);
}
